// DotProductAttention_14087492731337
// MI455X (gfx1250) — matrix operand order repaired and checked by reading, not compiled, not run on hardware
//
#include <hip/hip_runtime.h>

// ---------------------------------------------------------------------------
// Multi-head attention for MI455X (gfx1250): bf16 WMMA 16x16x32, fp32 accum.
// B=2, L=2048, D=1024, R=16 heads, d_r=64.
// ---------------------------------------------------------------------------

typedef __bf16 bf16;
typedef __attribute__((ext_vector_type(4)))  float v4f;
typedef __attribute__((ext_vector_type(8)))  float v8f;
typedef __attribute__((ext_vector_type(4)))  bf16  v4bf;
typedef __attribute__((ext_vector_type(8)))  bf16  v8bf;
typedef __attribute__((ext_vector_type(16))) bf16  v16bf;

#define WMMA_BF16(a, b, c) \
  __builtin_amdgcn_wmma_f32_16x16x32_bf16(false, (a), false, (b), (short)0, (c), false, false)

constexpr int BB = 2;     // batch
constexpr int LL = 2048;  // sequence
constexpr int DD = 1024;  // model dim
constexpr int RR = 16;    // heads
constexpr int DR = 64;    // head dim
constexpr int ML = BB * LL;  // 4096 rows for projections

// ---------------------------------------------------------------------------
// Async global->LDS staging (CDNA5 GLOBAL_LOAD_ASYNC_TO_LDS_B128, ASYNCcnt).
// Builtin signature (per toolchain diagnostic): param0 = int4 addrspace(1)*,
// so cast to exactly that; LDS side cast to int4 addrspace(3)*.
// ---------------------------------------------------------------------------
#if defined(__HIP_DEVICE_COMPILE__) && __has_builtin(__builtin_amdgcn_global_load_async_to_lds_b128)
#define HAVE_ASYNC_LDS 1
#else
#define HAVE_ASYNC_LDS 0
#endif

#if HAVE_ASYNC_LDS
typedef int v4i __attribute__((vector_size(16)));
typedef __attribute__((address_space(1))) v4i g_v4i;
typedef __attribute__((address_space(3))) v4i l_v4i;
__device__ __forceinline__ void copy16_g2lds(void* lds_dst, const void* gsrc) {
  __builtin_amdgcn_global_load_async_to_lds_b128((g_v4i*)gsrc, (l_v4i*)lds_dst, 0, 0);
}
__device__ __forceinline__ void wait_async_copies() {
  asm volatile("s_wait_asynccnt 0x0" ::: "memory");
}
#else
__device__ __forceinline__ void copy16_g2lds(void* lds_dst, const void* gsrc) {
  *(uint4*)lds_dst = *(const uint4*)gsrc;
}
__device__ __forceinline__ void wait_async_copies() {}
#endif

// ---------------------------------------------------------------------------
// fp32 -> bf16 elementwise convert (vector 4)
// ---------------------------------------------------------------------------
__global__ __launch_bounds__(256) void cvt_f32_bf16(const float* __restrict__ src,
                                                    bf16* __restrict__ dst, int n4) {
  int i = blockIdx.x * blockDim.x + threadIdx.x;
  if (i < n4) {
    v4f f = ((const v4f*)src)[i];
    ((v4bf*)dst)[i] = __builtin_convertvector(f, v4bf);
  }
}

// ---------------------------------------------------------------------------
// Fragment builders (CDNA5 WMMA 16x16x32 bf16 VGPR layouts, ISA 7.12.2)
// A 16x32: lane holds row m=lane&15; lanes<16 -> K {0..7,16..23}, lanes>=16 -> +8
// B 32x16: lane holds col n=lane&15; lanes<16 -> K {0..7,16..23}, lanes>=16 -> +8
// ---------------------------------------------------------------------------
__device__ __forceinline__ v16bf load_frag_a_bf16(const bf16* p, int lane, int ldk) {
  int m = lane & 15;
  int kb = (lane < 16) ? 0 : 8;
  v8bf lo = *(const v8bf*)(p + m * ldk + kb);
  v8bf hi = *(const v8bf*)(p + m * ldk + kb + 16);
  return __builtin_shufflevector(lo, hi, 0, 1, 2, 3, 4, 5, 6, 7, 8, 9, 10, 11, 12, 13, 14, 15);
}

__device__ __forceinline__ v16bf load_frag_b_bf16(const bf16* p, int lane, int ldn) {
  int n = lane & 15;
  int kb = (lane < 16) ? 0 : 8;
  v16bf r;
#pragma unroll
  for (int s = 0; s < 8; ++s) {
    r[s] = p[(kb + s) * ldn + n];
    r[8 + s] = p[(16 + kb + s) * ldn + n];
  }
  return r;
}

// ---------------------------------------------------------------------------
// Tiled bf16 GEMM: [4096 x 1024] @ [1024 x 1024].
// BM=BN=128, BK=32; 8 waves: waveM in 0..3 (32 rows), waveN in 0..1 (64 cols);
// each wave computes 32x64 = 8 WMMA per K-step from 2 A-frags + 4 B-frags.
// mode 0: heads layout [b, r, l, dr] bf16 with scale   (Q, V)
// mode 1: plain fp32 [m, n]                            (final Wc -> d_out)
// mode 2: transposed heads layout [b, r, dr, l] bf16   (K)
// ---------------------------------------------------------------------------
__global__ __launch_bounds__(256) void gemm_bf16_1024(const bf16* __restrict__ A,
                                                      const bf16* __restrict__ W,
                                                      bf16* __restrict__ outH,
                                                      float* __restrict__ outF,
                                                      int mode, float scale) {
  __shared__ bf16 As[128 * 32];  // 8 KB
  __shared__ bf16 Bs[32 * 128];  // 8 KB

  const int tid = threadIdx.x;
  const int lane = tid & 31, wave = tid >> 5;
  const int bm = blockIdx.y, bn = blockIdx.x;
  const int waveM = wave & 3, waveN = wave >> 2;

  v8f acc[2][4];
#pragma unroll
  for (int i = 0; i < 2; ++i)
#pragma unroll
    for (int t = 0; t < 4; ++t) acc[i][t] = (v8f){};

  for (int k0 = 0; k0 < 1024; k0 += 32) {
    // A tile 128x32 (8KB) + B tile 32x128 (8KB): 2 x 16B per thread each
#pragma unroll
    for (int i = 0; i < 2; ++i) {
      int e = tid + 256 * i;
      int arow = e >> 2, ac8 = (e & 3) * 8;
      copy16_g2lds(&As[arow * 32 + ac8],
                   &A[(size_t)(bm * 128 + arow) * 1024 + k0 + ac8]);
      int brow = e >> 4, bc8 = (e & 15) * 8;
      copy16_g2lds(&Bs[brow * 128 + bc8],
                   &W[(size_t)(k0 + brow) * 1024 + bn * 128 + bc8]);
    }
    if (k0 + 32 < 1024) {
      __builtin_prefetch(&A[(size_t)(bm * 128 + (tid >> 2)) * 1024 + k0 + 32], 0, 3);
      __builtin_prefetch(&W[(size_t)(k0 + 32 + (tid >> 4)) * 1024 + bn * 128], 0, 3);
    }
    wait_async_copies();
    __syncthreads();

    v16bf a0 = load_frag_a_bf16(&As[(waveM * 32 + 0) * 32], lane, 32);
    v16bf a1 = load_frag_a_bf16(&As[(waveM * 32 + 16) * 32], lane, 32);
#pragma unroll
    for (int t = 0; t < 4; ++t) {
      v16bf bt = load_frag_b_bf16(&Bs[waveN * 64 + 16 * t], lane, 128);
      acc[0][t] = WMMA_BF16(a0, bt, acc[0][t]);
      acc[1][t] = WMMA_BF16(a1, bt, acc[1][t]);
    }
    __syncthreads();
  }

  // epilogue: C layout m = 8*(lane>=16)+vgpr, n = lane&15
#pragma unroll
  for (int is = 0; is < 2; ++is) {
    const int gm0 = bm * 128 + waveM * 32 + is * 16 + 8 * (lane >> 4);
#pragma unroll
    for (int t = 0; t < 4; ++t) {
      const int n = bn * 128 + waveN * 64 + 16 * t + (lane & 15);
#pragma unroll
      for (int i = 0; i < 8; ++i) {
        int gm = gm0 + i;
        if (mode == 1) {
          outF[(size_t)gm * 1024 + n] = acc[is][t][i];
        } else {
          int bb = gm >> 11, l = gm & 2047;
          int r0 = n >> 6, d0 = n & 63;
          bf16 v = (bf16)(acc[is][t][i] * scale);
          if (mode == 0)
            outH[(((size_t)bb * RR + r0) * LL + l) * DR + d0] = v;   // [b,r,l,dr]
          else
            outH[(((size_t)bb * RR + r0) * DR + d0) * LL + l] = v;   // [b,r,dr,l]
        }
      }
    }
  }
}

// ---------------------------------------------------------------------------
// Fused attention per (b, head, 16-row q tile):
//   phase 1: S[16][2048] = Qh @ Kt (WMMA; K stored pre-transposed [dr][l])
//   phase 2: row softmax; fp32 attn -> d_out; rows compacted in place to bf16
//   phase 3: O[16][64] = P @ Vh (WMMA; all 8 waves, pairwise K-split + reduce)
//   epilogue: O -> bf16 concat buffer [b, l, r*64 + dr]
// LDS: 128KB scores + 16KB K/V chunk + 2KB Q = ~146KB of the 320KB/WGP.
// ---------------------------------------------------------------------------
__global__ __launch_bounds__(256) void attn_fused(const bf16* __restrict__ Qp,
                                                  const bf16* __restrict__ Kt,
                                                  const bf16* __restrict__ Vp,
                                                  float* __restrict__ attn_out,
                                                  bf16* __restrict__ concat_out) {
  __shared__ float S[16 * 2048];   // 128 KB scores; reused as bf16 P rows
  __shared__ bf16 KVs[128 * 64];   // 16 KB streaming K / V chunk (+ reduce scratch)
  __shared__ bf16 Qs[16 * 64];     // 2 KB Q tile

  const int tid = threadIdx.x;
  const int lane = tid & 31, wave = tid >> 5;
  const int qt = blockIdx.x, rr = blockIdx.y, b = blockIdx.z;

  const bf16* Qbase = Qp + (((size_t)b * RR + rr) * LL + qt * 16) * DR;
  const bf16* Ktbase = Kt + (((size_t)b * RR + rr) * DR) * LL;  // [64][2048]
  const bf16* Vbase = Vp + (((size_t)b * RR + rr) * LL) * DR;   // [2048][64]

  // Q tile: 16x64 bf16, 128 threads x 16B
  if (tid < 128) {
    int r0 = tid >> 3, c8 = (tid & 7) * 8;
    copy16_g2lds(&Qs[r0 * 64 + c8], &Qbase[(size_t)r0 * 64 + c8]);
  }
  wait_async_copies();
  __syncthreads();

  // Q fragments are loop-invariant: build once.
  const v16bf a0 = load_frag_a_bf16(&Qs[0], lane, 64);   // dr 0..31
  const v16bf a1 = load_frag_a_bf16(&Qs[32], lane, 64);  // dr 32..63

  // ---- phase 1: scores ----
  for (int kc = 0; kc < 16; ++kc) {
    // K chunk [64 dr][128 keys] straight from the pre-transposed layout.
#pragma unroll
    for (int i = 0; i < 4; ++i) {
      int e = tid + 256 * i;
      int row = e >> 4, c8 = (e & 15) * 8;
      copy16_g2lds(&KVs[row * 128 + c8],
                   &Ktbase[(size_t)row * LL + kc * 128 + c8]);
    }
    wait_async_copies();
    __syncthreads();

    // each wave: one 16-key tile of this chunk
    v8f acc = {};
    v16bf b0 = load_frag_b_bf16(&KVs[wave * 16], lane, 128);
    v16bf b1 = load_frag_b_bf16(&KVs[32 * 128 + wave * 16], lane, 128);
    acc = WMMA_BF16(a0, b0, acc);
    acc = WMMA_BF16(a1, b1, acc);

    int ncol = kc * 128 + wave * 16 + (lane & 15);
    int mrow = 8 * (lane >> 4);
#pragma unroll
    for (int i = 0; i < 8; ++i) S[(mrow + i) * 2048 + ncol] = acc[i];
    __syncthreads();
  }

  // ---- phase 2: softmax (2 rows per wave); fp32 attn to d_out; in-place
  //      bf16 compaction (write at byte 2j only clobbers indices <= j, all
  //      already consumed by this wave; rows are wave-private) ----
#pragma unroll
  for (int rep = 0; rep < 2; ++rep) {
    int row = wave * 2 + rep;
    float* srow = &S[row * 2048];
    bf16* prow = (bf16*)srow;

    float mx = -1e30f;
    for (int j = lane; j < 2048; j += 32) mx = fmaxf(mx, srow[j]);
#pragma unroll
    for (int off = 16; off > 0; off >>= 1) mx = fmaxf(mx, __shfl_xor(mx, off, 32));

    float sum = 0.f;
    for (int j = lane; j < 2048; j += 32) {
      float e = __expf(srow[j] - mx);
      srow[j] = e;
      sum += e;
    }
#pragma unroll
    for (int off = 16; off > 0; off >>= 1) sum += __shfl_xor(sum, off, 32);
    float inv = 1.f / sum;

    float* gout = attn_out + (((size_t)b * LL + (qt * 16 + row)) * RR + rr) * LL;
    for (int j = lane; j < 2048; j += 32) {
      float v = srow[j] * inv;
      gout[j] = v;
      prow[j] = (bf16)v;
    }
  }
  __syncthreads();

  // ---- phase 3: P @ V. All 8 waves: wave pairs (w, w+4) share a dr-ntile and
  //      split the 4 K-steps of each chunk; partials reduced via LDS. ----
  const bf16* Pbase = (const bf16*)S;  // rows stride 4096 bf16
  const int ntile = wave & 3;
  const int ksb = (wave < 4) ? 0 : 2;
  v8f oacc = {};
  for (int kc = 0; kc < 16; ++kc) {
    // V chunk [128 keys][64 dr] row-major (B rows = key = WMMA K dim)
#pragma unroll
    for (int i = 0; i < 4; ++i) {
      int e = tid + 256 * i;
      int row = e >> 3, c8 = (e & 7) * 8;
      copy16_g2lds(&KVs[row * 64 + c8],
                   &Vbase[(size_t)(kc * 128 + row) * 64 + c8]);
    }
    wait_async_copies();
    __syncthreads();
#pragma unroll
    for (int ks = 0; ks < 2; ++ks) {
      int kk = ksb + ks;
      v16bf pa = load_frag_a_bf16(Pbase + kc * 128 + kk * 32, lane, 4096);
      v16bf vb = load_frag_b_bf16(&KVs[(kk * 32) * 64 + ntile * 16], lane, 64);
      oacc = WMMA_BF16(pa, vb, oacc);
    }
    __syncthreads();
  }

  // pairwise reduction through LDS (reuse KV buffer: 4 ntiles x 16x16 fp32)
  float* red = (float*)KVs;
  const int mrow = 8 * (lane >> 4);
  if (wave >= 4) {
#pragma unroll
    for (int i = 0; i < 8; ++i)
      red[ntile * 256 + (mrow + i) * 16 + (lane & 15)] = oacc[i];
  }
  __syncthreads();
  if (wave < 4) {
    int ncol = rr * 64 + ntile * 16 + (lane & 15);
#pragma unroll
    for (int i = 0; i < 8; ++i) {
      float v = oacc[i] + red[ntile * 256 + (mrow + i) * 16 + (lane & 15)];
      int qq = qt * 16 + mrow + i;
      concat_out[((size_t)b * LL + qq) * DD + ncol] = (bf16)v;
    }
  }
}

// ---------------------------------------------------------------------------
// Host launcher
// ---------------------------------------------------------------------------
extern "C" void kernel_launch(void* const* d_in, const int* in_sizes, int n_in,
                              void* d_out, int out_size, void* d_ws, size_t ws_size,
                              hipStream_t stream) {
  const float* q = (const float*)d_in[0];
  const float* k = (const float*)d_in[1];
  const float* v = (const float*)d_in[2];
  const float* Wq = (const float*)d_in[3];
  const float* Wk = (const float*)d_in[4];
  const float* Wv = (const float*)d_in[5];
  const float* Wc = (const float*)d_in[6];

  float* att_v = (float*)d_out;                      // [B, L, D]
  float* att_groups = att_v + (size_t)BB * LL * DD;  // [B, L, R, L]

  const size_t actE = (size_t)ML * DD;  // 4096*1024 elements
  const size_t wE = (size_t)DD * DD;    // 1024*1024 elements
  bf16* p = (bf16*)d_ws;
  bf16* qb = p;   p += actE;
  bf16* kb = p;   p += actE;
  bf16* vb = p;   p += actE;
  bf16* Wqb = p;  p += wE;
  bf16* Wkb = p;  p += wE;
  bf16* Wvb = p;  p += wE;
  bf16* Wcb = p;  p += wE;
  bf16* Qp = p;   p += actE;  // [b, r, l, dr]
  bf16* Kp = p;   p += actE;  // [b, r, dr, l]  (transposed)
  bf16* Vp = p;   p += actE;  // [b, r, l, dr]
  bf16* cc = p;   p += actE;  // total 64 MB of workspace

  // 1) convert inputs and weights to bf16
  const int n4a = (int)(actE / 4), n4w = (int)(wE / 4);
  cvt_f32_bf16<<<n4a / 256, 256, 0, stream>>>(q, qb, n4a);
  cvt_f32_bf16<<<n4a / 256, 256, 0, stream>>>(k, kb, n4a);
  cvt_f32_bf16<<<n4a / 256, 256, 0, stream>>>(v, vb, n4a);
  cvt_f32_bf16<<<n4w / 256, 256, 0, stream>>>(Wq, Wqb, n4w);
  cvt_f32_bf16<<<n4w / 256, 256, 0, stream>>>(Wk, Wkb, n4w);
  cvt_f32_bf16<<<n4w / 256, 256, 0, stream>>>(Wv, Wvb, n4w);
  cvt_f32_bf16<<<n4w / 256, 256, 0, stream>>>(Wc, Wcb, n4w);

  // 2) Q/K/V projections (1/sqrt(d_r)=0.125 folded into Q; K pre-transposed)
  dim3 gg(DD / 128, ML / 128);
  gemm_bf16_1024<<<gg, 256, 0, stream>>>(qb, Wqb, Qp, nullptr, 0, 0.125f);
  gemm_bf16_1024<<<gg, 256, 0, stream>>>(kb, Wkb, Kp, nullptr, 2, 1.0f);
  gemm_bf16_1024<<<gg, 256, 0, stream>>>(vb, Wvb, Vp, nullptr, 0, 1.0f);

  // 3) fused attention (scores -> softmax -> attn@V), writes att_groups
  attn_fused<<<dim3(LL / 16, RR, BB), 256, 0, stream>>>(Qp, Kp, Vp, att_groups, cc);

  // 4) output projection -> att_v (fp32)
  gemm_bf16_1024<<<gg, 256, 0, stream>>>(cc, Wcb, nullptr, att_v, 1, 1.0f);
}
